// Conv2d_decompo_nbcp_baens_61942018343207
// MI455X (gfx1250) — matrix operand order repaired and checked by reading, not compiled, not run on hardware
//
#include <hip/hip_runtime.h>

// ---------------------------------------------------------------------------
// Types for CDNA5 WMMA (wave32, 16x16x32 bf16 -> f32)
// ---------------------------------------------------------------------------
typedef __attribute__((ext_vector_type(16))) __bf16 v16bf;
typedef __attribute__((ext_vector_type(8)))  __bf16 v8bf;
typedef __attribute__((ext_vector_type(8)))  float  v8f;

struct bfrag_pair { v8bf lo, hi; };

// round-to-nearest-even float -> bf16 (bit pattern)
static __device__ __forceinline__ unsigned short f2bf(float f) {
    unsigned u = __builtin_bit_cast(unsigned, f);
    unsigned r = u + 0x7FFFu + ((u >> 16) & 1u);
    return (unsigned short)(r >> 16);
}

// Problem constants
#define NENS 4
#define CIN  64
#define COUT 64
#define HW   160
#define DTOT (CIN * COUT * 9)   // 36864
#define KTOT (CIN * 9)          // 576 = 18 k-tiles of 32
#define NKT  18
#define PSTRIDE 72              // LDS stride (halves) per (row,col): bank spread + 16B align

// ---------------------------------------------------------------------------
// Kernel 1: decode low-bit weights -> bf16, scattered straight into WMMA
// A-fragment layout: off = ((((n*18 + kt)*4 + mtile)*32 + lane)*16 + elem)
// K ordering: k = (kh*3+kw)*64 + cin   (each 32-wide k-tile is one cin run)
// A 16x32 bf16 layout (ISA 7.12.2): lane = m + 16*h where h = kk bit3,
// elem = (kk&16 ? 8:0) + (kk&7).
// ---------------------------------------------------------------------------
__global__ __launch_bounds__(256)
void decode_weights(const float* __restrict__ U, const float* __restrict__ V,
                    const float* __restrict__ twopow,
                    const float* __restrict__ scale, const float* __restrict__ biasq,
                    unsigned short* __restrict__ wfrag) {
    int t = blockIdx.x * 256 + threadIdx.x;
    if (t >= NENS * DTOT) return;
    int n = t / DTOT;
    int d = t % DTOT;
    int o    = d / KTOT;          // cout
    int rem  = d % KTOT;
    int i    = rem / 9;           // cin
    int khkw = rem % 9;

    float u  = U[n * DTOT + d];
    float t0 = u * V[d];
    float t1 = u * V[DTOT + d];
    t0 = fminf(fmaxf(t0, -10.0f), 10.0f);
    t1 = fminf(fmaxf(t1, -10.0f), 10.0f);
    float s0 = 1.0f / (1.0f + __expf(-t0));
    float s1 = 1.0f / (1.0f + __expf(-t1));
    float integer = s0 * twopow[0] + s1 * twopow[1];
    float w = scale[n] * (integer - biasq[n] - 4.0f);

    int k   = khkw * CIN + i;
    int kt  = k >> 5;
    int kk  = k & 31;
    int m   = o & 15;
    int mt  = o >> 4;
    int h    = (kk >> 3) & 1;
    int elem = ((kk & 16) ? 8 : 0) + (kk & 7);
    int lane = m + (h << 4);
    size_t off = ((((size_t)n * NKT + kt) * 4 + mt) * 32 + lane) * 16 + elem;
    wfrag[off] = f2bf(w);
}

// ---------------------------------------------------------------------------
// Kernel 2: implicit-GEMM 3x3 conv via v_wmma_f32_16x16x32_bf16.
// Block: 128 threads = 4 waves. Tile: 64 couts x (4 rows x 16 cols) output.
// LDS patch: 6 rows x 18 cols x 64 cin (bf16, stride 72 halves).
// B 32x16 bf16 layout: lanes 0-15 hold K=0..7,16..23 (col = lane), lanes 16-31
// hold K=8..15,24..31 (col = lane-16); per-lane 16 halves = two 8-cin runs.
// Software pipeline: all 4 B fragments batched per K-tile (one DS clause,
// one wait, 4 back-to-back WMMAs); A fragment double-buffered one K-tile
// ahead via global_load_b128, prefetch two ahead.
// ---------------------------------------------------------------------------
__global__ __launch_bounds__(128)
void conv_baens(const float* __restrict__ x, const unsigned short* __restrict__ wfrag,
                const float* __restrict__ bias, float* __restrict__ out) {
    const int wt = blockIdx.x;       // 0..9   (w tiles of 16)
    const int ht = blockIdx.y;       // 0..39  (h tiles of 4)
    const int b  = blockIdx.z;       // 0..15  image
    const int n  = b & 3;            // ensemble member
    const int h0 = ht * 4, w0 = wt * 16;

    __shared__ unsigned short patch[6 * 18 * PSTRIDE];

    const int tid = threadIdx.x;
    const float* xb = x + (size_t)b * CIN * HW * HW;

    // ---- stage 6x18x64 fp32 patch -> bf16 LDS (zero-pad borders) ----
    for (int e = tid; e < 6 * 18 * CIN; e += 128) {
        int cc = e % 18;
        int t1 = e / 18;
        int rr = t1 % 6;
        int ci = t1 / 6;
        int gh = h0 - 1 + rr, gw = w0 - 1 + cc;
        float v = 0.0f;
        if ((unsigned)gh < (unsigned)HW && (unsigned)gw < (unsigned)HW)
            v = xb[((size_t)ci * HW + gh) * HW + gw];
        patch[(rr * 18 + cc) * PSTRIDE + ci] = f2bf(v);
    }
    __syncthreads();

    const int mt   = tid >> 5;       // wave id == m-tile (16 couts)
    const int lane = tid & 31;
    const int col  = lane & 15;      // output column within tile
    const int hi   = lane >> 4;      // lane half

    // accumulators initialized with bias (C operand of first WMMA)
    v8f acc[4];
    {
        const float* bp = bias + n * COUT + mt * 16 + hi * 8;
        v8f c;
#pragma unroll
        for (int r = 0; r < 8; ++r) c[r] = bp[r];
#pragma unroll
        for (int nt = 0; nt < 4; ++nt) acc[nt] = c;
    }

    // A fragment base for (n, mt): stride between k-tiles = 4*32*16 halves
    const unsigned short* abase =
        wfrag + (((size_t)n * NKT * 4 + mt) * 32 + lane) * 16;
    const size_t astride = (size_t)4 * 32 * 16;   // halves per k-tile step

    v16bf a_cur = *reinterpret_cast<const v16bf*>(abase);

#pragma unroll
    for (int kt = 0; kt < NKT; ++kt) {
        const int khkw = kt >> 1;
        const int kh = khkw / 3, kw = khkw % 3;
        const int i0 = (kt & 1) << 5;                    // cin base of k-tile
        const int kbase = i0 + (hi << 3);                // per-lane cin offset

        // ---- batch-load all 4 B fragments (8x ds_load_b128, one wait) ----
        v16bf bb[4];
#pragma unroll
        for (int nt = 0; nt < 4; ++nt) {
            const int rr = nt + kh;
            const int cc = col + kw;
            const unsigned short* p = &patch[(rr * 18 + cc) * PSTRIDE + kbase];
            v8bf blo = *reinterpret_cast<const v8bf*>(p);       // ds_load_b128
            v8bf bhi = *reinterpret_cast<const v8bf*>(p + 16);  // ds_load_b128
            bb[nt] = __builtin_bit_cast(v16bf, bfrag_pair{blo, bhi});
        }

        // ---- double-buffer next A fragment; prefetch one further ahead ----
        v16bf a_next = a_cur;
        if (kt + 1 < NKT) {
            a_next = *reinterpret_cast<const v16bf*>(abase + (size_t)(kt + 1) * astride);
            if (kt + 2 < NKT)   // global_prefetch_b8
                __builtin_prefetch(abase + (size_t)(kt + 2) * astride, 0, 3);
        }

        // ---- 4 back-to-back WMMAs on the XDL pipe ----
#pragma unroll
        for (int nt = 0; nt < 4; ++nt) {
            acc[nt] = __builtin_amdgcn_wmma_f32_16x16x32_bf16(
                false, a_cur, false, bb[nt], (short)0, acc[nt], false, false);
        }
        a_cur = a_next;
    }

    // ---- store D: lane col = w, VGPR r -> cout = mt*16 + r + 8*hi ----
#pragma unroll
    for (int nt = 0; nt < 4; ++nt) {
        const int oh = h0 + nt;
#pragma unroll
        for (int r = 0; r < 8; ++r) {
            const int co = mt * 16 + hi * 8 + r;
            out[(((size_t)b * COUT + co) * HW + oh) * HW + (w0 + col)] = acc[nt][r];
        }
    }
}

// ---------------------------------------------------------------------------
// Host launch
// ---------------------------------------------------------------------------
extern "C" void kernel_launch(void* const* d_in, const int* in_sizes, int n_in,
                              void* d_out, int out_size, void* d_ws, size_t ws_size,
                              hipStream_t stream) {
    const float* x      = (const float*)d_in[0];
    const float* U      = (const float*)d_in[1];
    const float* V      = (const float*)d_in[2];
    const float* twopow = (const float*)d_in[3];
    const float* scale  = (const float*)d_in[4];
    const float* biasq  = (const float*)d_in[5];
    const float* bias   = (const float*)d_in[6];
    float* out = (float*)d_out;
    unsigned short* wfrag = (unsigned short*)d_ws;   // 294912 B of scratch

    decode_weights<<<(NENS * DTOT + 255) / 256, 256, 0, stream>>>(
        U, V, twopow, scale, biasq, wfrag);

    dim3 grid(HW / 16, HW / 4, 16);   // (10, 40, 16)
    conv_baens<<<grid, 128, 0, stream>>>(x, wfrag, bias, out);
}
